// SA_Single_5368709120555
// MI455X (gfx1250) — matrix operand order repaired and checked by reading, not compiled, not run on hardware
//
#include <hip/hip_runtime.h>
#include <hip/hip_bf16.h>

// ---------------- problem constants (from reference) ----------------
#define B_   4
#define N_   8192
#define S_   2048      // NPOINT
#define K_   32        // neighbors
#define C_   64        // input feature channels
#define M_   (B_*S_*K_)  // 262144 GEMM rows
#define EPSB 1e-5f

typedef __bf16 v16bf __attribute__((ext_vector_type(16)));
typedef float  v8f   __attribute__((ext_vector_type(8)));

// =====================================================================
// 1) Farthest point sampling: one workgroup per batch, 1024 threads.
//    2048 sequential argmax steps; wave shuffle + LDS tree reduction.
// =====================================================================
__global__ __launch_bounds__(1024) void fps_kernel(const float* __restrict__ xyz,
                                                   int* __restrict__ fpsidx,
                                                   float* __restrict__ newxyz)
{
    const int b   = blockIdx.x;
    const int tid = threadIdx.x;
    const float* X = xyz + (size_t)b * N_ * 3;

    constexpr int PPT = N_ / 1024;  // 8 points per thread
    float px[PPT], py[PPT], pz[PPT], dmin[PPT];
#pragma unroll
    for (int i = 0; i < PPT; ++i) {
        int n = tid + i * 1024;
        px[i] = X[n*3+0]; py[i] = X[n*3+1]; pz[i] = X[n*3+2];
        dmin[i] = 1e10f;
    }

    __shared__ float s_val[32];
    __shared__ int   s_idx[32];
    __shared__ int   s_far;

    int far = 0;
    for (int s = 0; s < S_; ++s) {
        if (tid == 0) {
            fpsidx[b*S_ + s] = far;
            newxyz[((size_t)b*S_ + s)*3 + 0] = X[far*3+0];
            newxyz[((size_t)b*S_ + s)*3 + 1] = X[far*3+1];
            newxyz[((size_t)b*S_ + s)*3 + 2] = X[far*3+2];
        }
        const float cx = X[far*3+0], cy = X[far*3+1], cz = X[far*3+2];

        float bv = -1.0f; int bi = 0x7fffffff;
#pragma unroll
        for (int i = 0; i < PPT; ++i) {
            float dx = px[i]-cx, dy = py[i]-cy, dz = pz[i]-cz;
            float d  = dx*dx + dy*dy + dz*dz;
            dmin[i]  = fminf(dmin[i], d);
            if (dmin[i] > bv) { bv = dmin[i]; bi = tid + i*1024; }
        }
        // wave32 argmax (first-index tie-break like jnp.argmax)
#pragma unroll
        for (int off = 16; off > 0; off >>= 1) {
            float ov = __shfl_xor(bv, off);
            int   oi = __shfl_xor(bi, off);
            if (ov > bv || (ov == bv && oi < bi)) { bv = ov; bi = oi; }
        }
        const int wid = tid >> 5;
        if ((tid & 31) == 0) { s_val[wid] = bv; s_idx[wid] = bi; }
        __syncthreads();
        if (wid == 0) {
            bv = s_val[tid & 31]; bi = s_idx[tid & 31];
#pragma unroll
            for (int off = 16; off > 0; off >>= 1) {
                float ov = __shfl_xor(bv, off);
                int   oi = __shfl_xor(bi, off);
                if (ov > bv || (ov == bv && oi < bi)) { bv = ov; bi = oi; }
            }
            if (tid == 0) s_far = bi;
        }
        __syncthreads();
        far = s_far;
    }
}

// =====================================================================
// 2) KNN: one wave32 per centroid. The 32 current-best (dist,idx) pairs
//    live one per lane (K == wave width). Ballot-filtered replace-worst.
// =====================================================================
__global__ __launch_bounds__(256) void knn_kernel(const float* __restrict__ xyz,
                                                  const float* __restrict__ newxyz,
                                                  int* __restrict__ knn)
{
    const int lane  = threadIdx.x & 31;
    const int gwave = blockIdx.x * (blockDim.x >> 5) + (threadIdx.x >> 5);
    const int b = gwave / S_;
    const int s = gwave % S_;
    const float* X = xyz + (size_t)b * N_ * 3;

    const float cx = newxyz[((size_t)b*S_+s)*3+0];
    const float cy = newxyz[((size_t)b*S_+s)*3+1];
    const float cz = newxyz[((size_t)b*S_+s)*3+2];

    float kd = 1e30f;   // my kept distance
    int   ki = lane;    // my kept index

    for (int base = 0; base < N_; base += 32) {
        const int n = base + lane;
        float dx = X[n*3+0]-cx, dy = X[n*3+1]-cy, dz = X[n*3+2]-cz;
        float d  = dx*dx + dy*dy + dz*dz;

        // current worst kept distance across the wave
        float wmax = kd;
#pragma unroll
        for (int off = 16; off > 0; off >>= 1)
            wmax = fmaxf(wmax, __shfl_xor(wmax, off));

        unsigned long long mask = __ballot(d < wmax);
        while (mask) {
            int j = __ffsll(mask) - 1;
            mask &= mask - 1;
            float dj = __shfl(d, j);
            int   nj = base + j;
            // arg-max over kept distances -> lane holding the worst
            float bv = kd; int bl = lane;
#pragma unroll
            for (int off = 16; off > 0; off >>= 1) {
                float ov = __shfl_xor(bv, off);
                int   ol = __shfl_xor(bl, off);
                if (ov > bv) { bv = ov; bl = ol; }
            }
            if (dj < bv && lane == bl) { kd = dj; ki = nj; }
        }
    }
    knn[(size_t)(b*S_ + s)*K_ + lane] = ki;  // unordered set; max/BN are order-invariant
}

// =====================================================================
// 3) Gather + concat [rel_xyz(3) | feats(64) | zero pad -> 96] as bf16
// =====================================================================
__global__ void gather_feat(const float* __restrict__ xyz, const float* __restrict__ points,
                            const float* __restrict__ nx, const int* __restrict__ knn,
                            __hip_bfloat16* __restrict__ A0)
{
    size_t i = (size_t)blockIdx.x * blockDim.x + threadIdx.x;
    if (i >= (size_t)M_ * 96) return;
    int    c  = (int)(i % 96);
    size_t m  = i / 96;
    int    k  = (int)(m % K_);
    int    bs = (int)(m / K_);
    int    b  = bs / S_;
    int    n  = knn[(size_t)bs*K_ + k];
    float v;
    if (c < 3)       v = xyz[((size_t)b*N_ + n)*3 + c] - nx[(size_t)bs*3 + c];
    else if (c < 67) v = points[((size_t)b*C_ + (c-3))*N_ + n];
    else             v = 0.0f;
    A0[i] = __float2bfloat16(v);
}

// pack f32 weights (Cout x Kin) -> bf16 (Cout x Kp), zero padded
__global__ void pack_w(const float* __restrict__ w, __hip_bfloat16* __restrict__ wp,
                       int Cout, int Kin, int Kp)
{
    int i = blockIdx.x * blockDim.x + threadIdx.x;
    if (i >= Cout * Kp) return;
    int kk = i % Kp, o = i / Kp;
    wp[i] = __float2bfloat16(kk < Kin ? w[o*Kin + kk] : 0.0f);
}

__global__ void zero_f32(float* __restrict__ p, int n)
{
    int i = blockIdx.x * blockDim.x + threadIdx.x;
    if (i < n) p[i] = 0.0f;
}

// =====================================================================
// 4) bf16 WMMA GEMM, compile-time (Kd, Cout), 1x4 register blocking.
//    Block = 8 M-tiles x one 64-wide output strip. The strip's weights
//    (64 x Kd bf16, <=16KB) are staged once in LDS (padded pitch to
//    avoid 16-way bank conflicts on fragment reads), then B fragments
//    come from ds_load_b128 (short LDS latency, DScnt) while the A
//    stream stays pipelined on LOADcnt with a depth-2 ping/pong.
//    Fragment layouts per CDNA5 ISA:
//      A lane l<16 : row mBase+l,    K in {kc..kc+7, kc+16..kc+23}
//      A lane l>=16: row mBase+l-16, K in {kc+8..kc+15, kc+24..kc+31}
//      B lane l<16 : col cBase+l,    K in {kc..kc+7, kc+16..kc+23}
//      B lane l>=16: col cBase+l-16, K in {kc+8..kc+15, kc+24..kc+31}
//    Epilogue fuses per-channel sum/sumsq (BatchNorm stats) via f32 atomics.
// =====================================================================
union Frag { v16bf v; uint4 u[2]; };

template<int Kd, int Cout>
__global__ __launch_bounds__(256) void gemm_wmma_bf16(
    const __hip_bfloat16* __restrict__ A,   // M x Kd (Kd multiple of 32)
    const __hip_bfloat16* __restrict__ W,   // Cout x Kd row-major
    float* __restrict__ out,                // M x Cout
    float* __restrict__ sum, float* __restrict__ sumsq)
{
    constexpr int NK     = Kd / 32;         // K steps
    constexpr int strips = Cout / 64;       // 64-wide output strips
    constexpr int PITCH  = Kd + 8;          // LDS row pitch in halfs (16B pad: stagger banks)
    constexpr int ROWV   = Kd / 8;          // uint4 per weight row

    __shared__ unsigned short sw[64 * PITCH];

    const int lane  = threadIdx.x & 31;
    const int widx  = threadIdx.x >> 5;
    const int bm    = blockIdx.x / strips;
    const int cBase = (blockIdx.x % strips) * 64;
    const int tm    = bm * 8 + widx;

    // cooperative stage of this block's 64 x Kd weight strip into LDS
    for (int i = threadIdx.x; i < 64 * ROWV; i += 256) {
        const int r = i / ROWV, c0 = i % ROWV;
        *reinterpret_cast<uint4*>(&sw[r * PITCH + c0 * 8]) =
            reinterpret_cast<const uint4*>(W + (size_t)(cBase + r) * Kd)[c0];
    }
    __syncthreads();

    const __hip_bfloat16* arow = A + (size_t)(tm*16 + (lane & 15)) * Kd + ((lane & 16) ? 8 : 0);
    const int swBase = (lane & 15) * PITCH + ((lane & 16) ? 8 : 0);

    Frag fa[2];

    auto load_a = [&](int s, int buf) {
        const int kc = s * 32;
        fa[buf].u[0] = *reinterpret_cast<const uint4*>(arow + kc);
        fa[buf].u[1] = *reinterpret_cast<const uint4*>(arow + kc + 16);
    };

    v8f acc[4] = {v8f{}, v8f{}, v8f{}, v8f{}};
    load_a(0, 0);
#pragma unroll
    for (int s = 0; s < NK; ++s) {
        const int cur = s & 1;
        if (s + 1 < NK) {
            __builtin_prefetch(arow + (s + 2) * 32, 0, 1);   // global_prefetch_b8
            load_a(s + 1, cur ^ 1);
        }
        const int kc = s * 32;
#pragma unroll
        for (int t = 0; t < 4; ++t) {
            Frag fb;   // ds_load_b128 x2 from the staged strip
            fb.u[0] = *reinterpret_cast<const uint4*>(&sw[swBase + t*16*PITCH + kc]);
            fb.u[1] = *reinterpret_cast<const uint4*>(&sw[swBase + t*16*PITCH + kc + 16]);
            acc[t] = __builtin_amdgcn_wmma_f32_16x16x32_bf16(false, fa[cur].v, false, fb.v,
                                                             (short)0, acc[t], false, false);
        }
    }

    // C/D layout: lanes 0-15 -> M = r, lanes 16-31 -> M = 8+r, N = lane&15
    const int mrow0 = tm*16 + ((lane & 16) ? 8 : 0);
#pragma unroll
    for (int t = 0; t < 4; ++t) {
        const int col = cBase + t*16 + (lane & 15);
        float s = 0.0f, s2 = 0.0f;
#pragma unroll
        for (int r = 0; r < 8; ++r) {
            float v = acc[t][r];
            out[(size_t)(mrow0 + r) * Cout + col] = v;
            s += v; s2 += v * v;
        }
        atomicAdd(&sum[col],   s);    // global_atomic_add_f32
        atomicAdd(&sumsq[col], s2);
    }
}

// training-mode BN (per-channel over all M rows) + ReLU, repack bf16
__global__ void bn_relu_pack(const float* __restrict__ x, const float* __restrict__ sum,
                             const float* __restrict__ sumsq, const float* __restrict__ g,
                             const float* __restrict__ bta, __hip_bfloat16* __restrict__ y,
                             int Cout)
{
    size_t i = (size_t)blockIdx.x * blockDim.x + threadIdx.x;
    if (i >= (size_t)M_ * (size_t)Cout) return;
    int   c    = (int)(i % Cout);
    float inv  = 1.0f / (float)M_;
    float mean = sum[c] * inv;
    float var  = sumsq[c] * inv - mean * mean;
    float sc   = rsqrtf(var + EPSB) * g[c];
    float sh   = bta[c] - mean * sc;
    float v    = x[i] * sc + sh;
    y[i] = __float2bfloat16(v > 0.0f ? v : 0.0f);
}

// last layer: BN + ReLU + max over K neighbors + transpose to (B,256,S)
__global__ void final_max(const float* __restrict__ x, const float* __restrict__ sum,
                          const float* __restrict__ sumsq, const float* __restrict__ g,
                          const float* __restrict__ bta, float* __restrict__ outp)
{
    int i = blockIdx.x * blockDim.x + threadIdx.x;   // over B*S*256
    if (i >= B_*S_*256) return;
    int c  = i % 256;
    int bs = i / 256;
    int s  = bs % S_;
    int b  = bs / S_;
    float inv  = 1.0f / (float)M_;
    float mean = sum[c] * inv;
    float var  = sumsq[c] * inv - mean * mean;
    float sc   = rsqrtf(var + EPSB) * g[c];
    float sh   = bta[c] - mean * sc;

    const float* xp = x + (size_t)bs * K_ * 256 + c;
    float m = -1e30f;
#pragma unroll 4
    for (int k = 0; k < K_; ++k)
        m = fmaxf(m, xp[(size_t)k * 256] * sc + sh);
    outp[((size_t)b*256 + c)*S_ + s] = fmaxf(m, 0.0f);
}

// =====================================================================
// launch
// =====================================================================
extern "C" void kernel_launch(void* const* d_in, const int* in_sizes, int n_in,
                              void* d_out, int out_size, void* d_ws, size_t ws_size,
                              hipStream_t stream)
{
    const float* xyz    = (const float*)d_in[0];
    const float* points = (const float*)d_in[1];
    const float* w0 = (const float*)d_in[2];
    const float* g0 = (const float*)d_in[3];
    const float* b0 = (const float*)d_in[4];
    const float* w1 = (const float*)d_in[5];
    const float* g1 = (const float*)d_in[6];
    const float* b1 = (const float*)d_in[7];
    const float* w2 = (const float*)d_in[8];
    const float* g2 = (const float*)d_in[9];
    const float* b2 = (const float*)d_in[10];

    float* out    = (float*)d_out;
    float* newxyz = out;                          // (B,S,3)
    float* newpts = out + (size_t)B_ * S_ * 3;    // (B,256,S)

    char*  ws  = (char*)d_ws;
    size_t off = 0;
    auto carve = [&](size_t bytes) -> void* {
        void* p = ws + off;
        off += (bytes + 255) & ~(size_t)255;
        return p;
    };
    int*            fpsidx = (int*)carve((size_t)B_*S_*4);
    int*            knn    = (int*)carve((size_t)B_*S_*K_*4);
    float*          stats  = (float*)carve((size_t)1536*4);
    __hip_bfloat16* wp0    = (__hip_bfloat16*)carve((size_t)64 * 96 * 2);
    __hip_bfloat16* wp1    = (__hip_bfloat16*)carve((size_t)128* 64 * 2);
    __hip_bfloat16* wp2    = (__hip_bfloat16*)carve((size_t)256*128 * 2);
    __hip_bfloat16* act    = (__hip_bfloat16*)carve((size_t)M_*128*2);  // widest act (96/64/128)
    float*          outF   = (float*)carve((size_t)M_*256*4);           // widest GEMM out

    float* sum0 = stats + 0,    *sq0 = stats + 256;
    float* sum1 = stats + 512,  *sq1 = stats + 768;
    float* sum2 = stats + 1024, *sq2 = stats + 1280;

    zero_f32<<<6, 256, 0, stream>>>(stats, 1536);
    pack_w<<<(64*96   + 255)/256, 256, 0, stream>>>(w0, wp0, 64,  67,  96);
    pack_w<<<(128*64  + 255)/256, 256, 0, stream>>>(w1, wp1, 128, 64,  64);
    pack_w<<<(256*128 + 255)/256, 256, 0, stream>>>(w2, wp2, 256, 128, 128);

    fps_kernel<<<B_, 1024, 0, stream>>>(xyz, fpsidx, newxyz);
    knn_kernel<<<(B_*S_)/8, 256, 0, stream>>>(xyz, newxyz, knn);

    unsigned gblocks = (unsigned)(((size_t)M_*96 + 255) / 256);
    gather_feat<<<gblocks, 256, 0, stream>>>(xyz, points, newxyz, knn, act);

    // blocks = (M/16/8) * (Cout/64); 8 waves (M-tiles) per block, one strip per block
    // layer 0: Kd=96, Cout=64
    gemm_wmma_bf16<96,64><<<(M_/16/8)*(64/64),  256, 0, stream>>>(act, wp0, outF, sum0, sq0);
    bn_relu_pack<<<(unsigned)(((size_t)M_*64 + 255)/256), 256, 0, stream>>>(outF, sum0, sq0, g0, b0, act, 64);

    // layer 1: Kd=64, Cout=128
    gemm_wmma_bf16<64,128><<<(M_/16/8)*(128/64), 256, 0, stream>>>(act, wp1, outF, sum1, sq1);
    bn_relu_pack<<<(unsigned)(((size_t)M_*128 + 255)/256), 256, 0, stream>>>(outF, sum1, sq1, g1, b1, act, 128);

    // layer 2: Kd=128, Cout=256
    gemm_wmma_bf16<128,256><<<(M_/16/8)*(256/64), 256, 0, stream>>>(act, wp2, outF, sum2, sq2);

    final_max<<<(B_*S_*256)/256, 256, 0, stream>>>(outF, sum2, sq2, g2, b2, newpts);
}
